// Pct_3DTA_2826088481366
// MI455X (gfx1250) — matrix operand order repaired and checked by reading, not compiled, not run on hardware
//
#include <hip/hip_runtime.h>
#include <hip/hip_bf16.h>

// ============================================================================
// PCT (Point Cloud Transformer) forward for MI455X (gfx1250, wave32).
// All 1x1 convs / attention einsums run through v_wmma_f32_16x16x32_f16
// (f16 activations+weights, f32 accumulate). BN is training-mode batch stats:
// two-pass (rowwise mean/var reduce, then fused normalize+activation+residual).
//
// GEMM: 64x64 block tile, 8 waves, 2 WMMA accumulators per wave. A/B tiles are
// staged in LDS in *fragment-major* layout (ISA 7.12.2 VGPR layouts), so each
// lane's 16 f16 fragment elements are contiguous -> fragment load = 2x
// ds_load_b128. A-tile staging (row-major case) uses the CDNA5 async copy
// engine: global_load_async_to_lds_b128 (ASYNCcnt) so the data never touches
// VGPRs; B-tile staging lane-reshuffles, so it stays global_load_b128 +
// ds_store scatter. All dims tile-aligned (conv1 is K-padded 6->32): no
// bounds guards anywhere in the hot loop.
//
// Input index map (assumes insertion-order flattening of setup_inputs dict):
//   0 x | 1 conv1_w 2 bn1_g 3 bn1_b | 4 conv2_w 5 bn2_g 6 bn2_b
//   7..12  local0 {w1,g1,b1,w2,g2,b2}
//   13..18 local1 {w1,g1,b1,w2,g2,b2}
//   19 fuse1_w 20 fuse1_g 21 fuse1_b
//   22..27 tr.w[0..5]  28..33 tr.g[0..5]  34..39 tr.b[0..5]
//   40+7j .. : sa[j] {qk_w, v_w, v_b, t_w, t_b, bn_g, bn_b}, j=0..7
//   96 fuse2_w 97 fuse2_g 98 fuse2_b
//   99 l1_w 100 bn6_g 101 bn6_b | 102 l2_w 103 bn7_g 104 bn7_b
//   105 l3_w 106 l3_b
// ============================================================================

typedef _Float16 half_t;
typedef __attribute__((ext_vector_type(16))) _Float16 v16h;
typedef __attribute__((ext_vector_type(8)))  _Float16 v8h;
typedef __attribute__((ext_vector_type(8)))  float    v8f;

static constexpr int BV  = 16;    // batch
static constexpr int NV  = 8192;  // points
static constexpr int KNN = 32;    // neighbors

// ---------------------------------------------------------------------------
// WMMA GEMM:  C(M,N) = op(A)(MxK) * B(KxN)
//   A,B f16. tA: A[m,k]=Ap[k*lda+m] else Ap[m*lda+k].
//   Output f32 (Cf) or f16 (Ch) — exactly one non-null.
//   Batched over gridDim.z with element strides sA/sB/sC.
// Requirements (guaranteed by host): M%64==0, N%64==0, K%32==0,
// lda%8==0, ldb%8==0, batch strides keep 16B alignment.
// ---------------------------------------------------------------------------
static constexpr int TBM = 64, TBN = 64, TBK = 32;

__global__ __launch_bounds__(256)
void k_gemm_wmma(const half_t* __restrict__ A, const half_t* __restrict__ Bp,
                 float* __restrict__ Cf, half_t* __restrict__ Ch,
                 int M, int N, int Kd, int lda, int ldb, int ldc,
                 int tA, long long sA, long long sB, long long sC)
{
  // Fragment-major tiles: [tile][lane][element], lane's 16 halves contiguous.
  __shared__ half_t AsF[4][32][16];   // 4 m-tiles of 16x32   (4 KB)
  __shared__ half_t BsF[4][32][16];   // 4 n-tiles of 32x16   (4 KB)

  const int bz = blockIdx.z;
  A  += (long long)bz * sA;
  Bp += (long long)bz * sB;
  if (Cf) Cf += (long long)bz * sC;
  if (Ch) Ch += (long long)bz * sC;

  const int m0 = blockIdx.y * TBM;
  const int n0 = blockIdx.x * TBN;
  const int tid  = threadIdx.x;
  const int wave = tid >> 5, lane = tid & 31;
  const int wm = wave >> 1, wn = wave & 1;   // wave: m-tile 0..3, n-pair 0..1
  const int r  = lane & 15, hi = lane >> 4;

  // A staging (tA=0): thread -> (m-tile, row, 8-half k-chunk); 1:1 async copy.
  const int a_mt = tid >> 6, a_r = (tid >> 2) & 15, a_c = tid & 3;
  // LDS byte offset for this thread's A-fragment slot (flat addr low 32 bits
  // are the LDS offset per ISA 10.2 aperture mapping).
  const unsigned a_lds =
      (unsigned)(size_t)&AsF[a_mt][a_r + 16 * (a_c & 1)][(a_c >> 1) * 8];
  // A staging (tA=1): thread -> (k row, 8-half m-chunk); scatter 8 halves.
  const int at_k = tid >> 3, at_mc = tid & 7;
  // B staging: thread -> (k row, 8-half n-chunk); scatter 8 halves.
  const int b_k = tid >> 3, b_nc = tid & 7;

  v8f acc0 = {}, acc1 = {};

  for (int k0 = 0; k0 < Kd; k0 += TBK) {
    if (!tA) {
      // A row-major: 8 consecutive k of one row land contiguously in the
      // fragment slot -> one async 16B global->LDS copy, no VGPR bounce.
      const half_t* gp = A + (long long)(m0 + a_mt * 16 + a_r) * lda + (k0 + 8 * a_c);
      if (k0 + TBK < Kd) __builtin_prefetch((const void*)(gp + TBK), 0, 1);
      asm volatile("global_load_async_to_lds_b128 %0, %1, off"
                   :: "v"(a_lds), "v"(gp) : "memory");
    } else {
      // A col-major (energy): contiguous along m; scatter into 8 lanes' blocks.
      const half_t* gp = A + (long long)(k0 + at_k) * lda + (m0 + 8 * at_mc);
      v8h av = *(const v8h*)gp;
      if (k0 + TBK < Kd) __builtin_prefetch((const void*)(gp + (long long)TBK * lda), 0, 1);
      const int e  = ((at_k & 16) ? 8 : 0) + (at_k & 7);
      const int lh = ((at_k & 15) >= 8) ? 16 : 0;
#pragma unroll
      for (int u = 0; u < 8; ++u) {
        int m = 8 * at_mc + u;
        AsF[m >> 4][(m & 15) + lh][e] = av[u];
      }
    }
    {
      // B row-major: contiguous along n; scatter into 8 lanes' blocks.
      const half_t* gp = Bp + (long long)(k0 + b_k) * ldb + (n0 + 8 * b_nc);
      v8h bv = *(const v8h*)gp;
      if (k0 + TBK < Kd) __builtin_prefetch((const void*)(gp + (long long)TBK * ldb), 0, 1);
      const int e  = ((b_k & 16) ? 8 : 0) + (b_k & 7);
      const int kh = ((b_k & 15) >= 8) ? 16 : 0;
#pragma unroll
      for (int u = 0; u < 8; ++u) {
        int n = 8 * b_nc + u;
        BsF[n >> 4][(n & 15) + kh][e] = bv[u];
      }
    }
    // Drain the async copies before the tile barrier (ASYNCcnt), then sync.
    asm volatile("s_wait_asynccnt 0" ::: "memory");
    __syncthreads();

    v16h af = *(const v16h*)&AsF[wm][lane][0];          // 2x ds_load_b128
    v16h b0 = *(const v16h*)&BsF[2 * wn][lane][0];
    v16h b1 = *(const v16h*)&BsF[2 * wn + 1][lane][0];
    acc0 = __builtin_amdgcn_wmma_f32_16x16x32_f16(false, af, false, b0,
                                                  (short)0, acc0, false, false);
    acc1 = __builtin_amdgcn_wmma_f32_16x16x32_f16(false, af, false, b1,
                                                  (short)0, acc1, false, false);
    __syncthreads();
  }

  // C layout: VGPR v, lanes 0-15 -> M=v, lanes 16-31 -> M=v+8; N=lane&15
#pragma unroll
  for (int v = 0; v < 8; ++v) {
    int gm  = m0 + wm * 16 + hi * 8 + v;
    int gn0 = n0 + (2 * wn) * 16 + r;
    int gn1 = n0 + (2 * wn + 1) * 16 + r;
    if (Cf) {
      Cf[(long long)gm * ldc + gn0] = acc0[v];
      Cf[(long long)gm * ldc + gn1] = acc1[v];
    }
    if (Ch) {
      Ch[(long long)gm * ldc + gn0] = (half_t)acc0[v];
      Ch[(long long)gm * ldc + gn1] = (half_t)acc1[v];
    }
  }
}

// ---------------------------------------------------------------------------
// Rowwise batch stats over (C, P) f16 matrix: mean[c], var[c] (population)
// ---------------------------------------------------------------------------
__global__ __launch_bounds__(256)
void k_rowstats(const half_t* __restrict__ X, float* __restrict__ mean,
                float* __restrict__ var, long long P)
{
  __shared__ float rs[256], rq[256];
  const int c = blockIdx.x, t = threadIdx.x;
  const half_t* row = X + (long long)c * P;
  float s = 0.f, q = 0.f;
  for (long long i = t; i < P; i += 256) {
    float v = (float)row[i];
    s += v; q += v * v;
  }
  rs[t] = s; rq[t] = q;
  __syncthreads();
  for (int o = 128; o > 0; o >>= 1) {
    if (t < o) { rs[t] += rs[t + o]; rq[t] += rq[t + o]; }
    __syncthreads();
  }
  if (t == 0) {
    float m = rs[0] / (float)P;
    mean[c] = m;
    var[c]  = rq[0] / (float)P - m * m;
  }
}

// dst = (res ? res + : ) act( (x-mean)*rsqrt(var+eps)*g + b ) ; act 0/1/2=none/relu/leaky0.2
__global__ __launch_bounds__(256)
void k_bn_act(const half_t* __restrict__ X, const float* __restrict__ mean,
              const float* __restrict__ var, const float* __restrict__ g,
              const float* __restrict__ bb, const half_t* __restrict__ res,
              half_t* __restrict__ dst, long long P, int C, int act)
{
  long long i = (long long)blockIdx.x * 256 + threadIdx.x;
  if (i >= (long long)C * P) return;
  int c = (int)(i / P);
  float y = ((float)X[i] - mean[c]) * rsqrtf(var[c] + 1e-5f) * g[c] + bb[c];
  if (act == 1)      y = fmaxf(y, 0.f);
  else if (act == 2) y = (y > 0.f) ? y : 0.2f * y;
  if (res) y += (float)res[i];
  dst[i] = (half_t)y;
}

// ---------------------------------------------------------------------------
// Small elementwise / packing kernels
// ---------------------------------------------------------------------------
__global__ void k_f32_to_f16(const float* __restrict__ s, half_t* __restrict__ d,
                             long long n)
{
  long long i = (long long)blockIdx.x * 256 + threadIdx.x;
  if (i < n) d[i] = (half_t)s[i];
}

// zero-padded weight convert: dst (O,Kp) from src (O,Ks), Kp>=Ks
__global__ void k_pad_w(const float* __restrict__ s, half_t* __restrict__ d,
                        int O, int Ks, int Kp)
{
  int i = blockIdx.x * 256 + threadIdx.x;
  if (i >= O * Kp) return;
  int o = i / Kp, k = i % Kp;
  d[i] = (half_t)(k < Ks ? s[o * Ks + k] : 0.f);
}

// pack x (B,6,N) -> xyz (B,N,3) f32 and Ain (32, B*N) f16 (rows 6..31 zero)
__global__ void k_pack_input(const float* __restrict__ x, float* __restrict__ xyz,
                             half_t* __restrict__ Ain)
{
  long long i = (long long)blockIdx.x * 256 + threadIdx.x;   // over B*N
  if (i >= (long long)BV * NV) return;
  int b = (int)(i / NV), n = (int)(i % NV);
#pragma unroll
  for (int c = 0; c < 6; ++c) {
    float v = x[((long long)b * 6 + c) * NV + n];
    Ain[(long long)c * (BV * NV) + i] = (half_t)v;
    if (c < 3) xyz[i * 3 + c] = v;
  }
#pragma unroll
  for (int c = 6; c < 32; ++c)
    Ain[(long long)c * (BV * NV) + i] = (half_t)0.0f;
}

__global__ void k_bias(half_t* __restrict__ X, const float* __restrict__ bias,
                       long long P, int C)
{
  long long i = (long long)blockIdx.x * 256 + threadIdx.x;
  if (i >= (long long)C * P) return;
  int c = (int)(i / P);
  X[i] = (half_t)((float)X[i] + bias[c]);
}

__global__ void k_sub(half_t* __restrict__ d, const half_t* __restrict__ a,
                      const half_t* __restrict__ b, long long n)
{
  long long i = (long long)blockIdx.x * 256 + threadIdx.x;
  if (i < n) d[i] = (half_t)((float)a[i] - (float)b[i]);
}

__global__ void k_add(half_t* __restrict__ d, const half_t* __restrict__ a,
                      const half_t* __restrict__ b, long long n)
{
  long long i = (long long)blockIdx.x * 256 + threadIdx.x;
  if (i < n) d[i] = (half_t)((float)a[i] + (float)b[i]);
}

__global__ void k_copy_h(half_t* __restrict__ d, const half_t* __restrict__ s,
                         long long n)
{
  long long i = (long long)blockIdx.x * 256 + threadIdx.x;
  if (i < n) d[i] = s[i];
}

// hs[c, b*256+j] = h[c, b*8192 + 32*j]
__global__ void k_stride_cols(const half_t* __restrict__ h, half_t* __restrict__ hs)
{
  long long i = (long long)blockIdx.x * 256 + threadIdx.x;  // over 64*BV*256
  if (i >= 64LL * BV * 256) return;
  int c = (int)(i / (BV * 256));
  int rem = (int)(i % (BV * 256));
  int b = rem >> 8, j = rem & 255;
  hs[i] = h[(long long)c * (BV * NV) + (long long)b * NV + 32 * j];
}

// ---------------------------------------------------------------------------
// Farthest point sampling: one workgroup per batch; dist[] resident in LDS.
// Records current 'far' BEFORE updating (matches lax.scan semantics).
// ---------------------------------------------------------------------------
__global__ __launch_bounds__(256)
void k_fps(const float* __restrict__ xyz, int* __restrict__ outidx, int Np, int S)
{
  __shared__ float dist[8192];
  __shared__ float rmax[256];
  __shared__ int   rarg[256];
  __shared__ int   s_far;
  const int b = blockIdx.x, t = threadIdx.x;
  const float* X = xyz + (long long)b * Np * 3;
  for (int i = t; i < Np; i += 256) dist[i] = 1e10f;
  if (t == 0) s_far = 0;
  __syncthreads();
  for (int s = 0; s < S; ++s) {
    int far = s_far;
    if (t == 0) outidx[b * S + s] = far;
    float cx = X[far * 3], cy = X[far * 3 + 1], cz = X[far * 3 + 2];
    float bm = -1.f; int ba = 0;
    for (int i = t; i < Np; i += 256) {
      float dx = X[i * 3] - cx, dy = X[i * 3 + 1] - cy, dz = X[i * 3 + 2] - cz;
      float nd = fminf(dist[i], dx * dx + dy * dy + dz * dz);
      dist[i] = nd;
      if (nd > bm) { bm = nd; ba = i; }
    }
    rmax[t] = bm; rarg[t] = ba;
    __syncthreads();
    for (int o = 128; o > 0; o >>= 1) {
      if (t < o) {
        if (rmax[t + o] > rmax[t] ||
            (rmax[t + o] == rmax[t] && rarg[t + o] < rarg[t])) {
          rmax[t] = rmax[t + o]; rarg[t] = rarg[t + o];
        }
      }
      __syncthreads();
    }
    if (t == 0) s_far = rarg[0];
    __syncthreads();
  }
}

__global__ void k_gather_xyz(const float* __restrict__ src, const int* __restrict__ idx,
                             float* __restrict__ dst, int Np, int S)
{
  int i = blockIdx.x * 256 + threadIdx.x;  // over BV*S
  if (i >= BV * S) return;
  int b = i / S;
  int p = idx[i];
#pragma unroll
  for (int c = 0; c < 3; ++c)
    dst[i * 3 + c] = src[((long long)b * Np + p) * 3 + c];
}

// kNN: one workgroup per (b,center); d2[] in LDS; 32 sequential argmin rounds.
__global__ __launch_bounds__(256)
void k_knn(const float* __restrict__ xyz, const float* __restrict__ cxyz,
           int* __restrict__ nidx, int Np, int S, int Kn)
{
  __shared__ float d2[8192];
  __shared__ float rmin[256];
  __shared__ int   rarg[256];
  const int g = blockIdx.x, t = threadIdx.x;
  const int b = g / S;
  const float* X = xyz + (long long)b * Np * 3;
  float cx = cxyz[g * 3], cy = cxyz[g * 3 + 1], cz = cxyz[g * 3 + 2];
  for (int i = t; i < Np; i += 256) {
    float dx = X[i * 3] - cx, dy = X[i * 3 + 1] - cy, dz = X[i * 3 + 2] - cz;
    d2[i] = dx * dx + dy * dy + dz * dz;
  }
  __syncthreads();
  for (int j = 0; j < Kn; ++j) {
    float bm = 1e30f; int ba = 0;
    for (int i = t; i < Np; i += 256) {
      float v = d2[i];
      if (v < bm) { bm = v; ba = i; }
    }
    rmin[t] = bm; rarg[t] = ba;
    __syncthreads();
    for (int o = 128; o > 0; o >>= 1) {
      if (t < o) {
        if (rmin[t + o] < rmin[t] ||
            (rmin[t + o] == rmin[t] && rarg[t + o] < rarg[t])) {
          rmin[t] = rmin[t + o]; rarg[t] = rarg[t + o];
        }
      }
      __syncthreads();
    }
    if (t == 0) { nidx[(long long)g * Kn + j] = rarg[0]; d2[rarg[0]] = 1e30f; }
    __syncthreads();
  }
}

// Build grouped features: rows 0..D-1 = neighbor - center, rows D..2D-1 = center.
// feat: (D, BV*Np) f16 ; output Xo: (2D, BV*S*Kn) f16, col = (b*S+s)*Kn + j
__global__ void k_group(const half_t* __restrict__ feat, const int* __restrict__ fpsidx,
                        const int* __restrict__ nidx, half_t* __restrict__ Xo,
                        int D, int Np, int S, int Kn)
{
  long long col = (long long)blockIdx.x * 256 + threadIdx.x;
  long long tot = (long long)BV * S * Kn;
  if (col >= tot) return;
  int b = (int)(col / ((long long)S * Kn));
  int rem = (int)(col % ((long long)S * Kn));
  int s = rem / Kn, j = rem % Kn;
  int nb = nidx[((long long)(b * S + s)) * Kn + j];
  int ci = fpsidx[b * S + s];
  long long ldf = (long long)BV * Np;
  for (int c = 0; c < D; ++c) {
    half_t gv = feat[(long long)c * ldf + (long long)b * Np + nb];
    half_t cv = feat[(long long)c * ldf + (long long)b * Np + ci];
    Xo[(long long)c * tot + col]       = (half_t)((float)gv - (float)cv);
    Xo[(long long)(c + D) * tot + col] = cv;
  }
}

// max over Kn consecutive columns: (C, Pout*Kn) -> (C, Pout)
__global__ void k_maxpool(const half_t* __restrict__ X, half_t* __restrict__ out,
                          long long Pout, int Kn, int C)
{
  long long i = (long long)blockIdx.x * 256 + threadIdx.x;
  if (i >= (long long)C * Pout) return;
  int c = (int)(i / Pout);
  long long col = i % Pout;
  const half_t* s = X + ((long long)c * Pout + col) * Kn;
  float m = (float)s[0];
  for (int j = 1; j < Kn; ++j) m = fmaxf(m, (float)s[j]);
  out[i] = (half_t)m;
}

// ---------------------------------------------------------------------------
// Attention helpers (energy layout: rows n (256), cols b*256+m, ld 4096, f32)
// ---------------------------------------------------------------------------
__global__ __launch_bounds__(256)
void k_softmax_rows(float* __restrict__ E)
{
  __shared__ float red[256];
  const int g = blockIdx.x;            // b*256 + n
  const int b = g >> 8, n = g & 255, t = threadIdx.x;
  float* row = E + (long long)n * 4096 + b * 256;
  float v = row[t];
  red[t] = v; __syncthreads();
  for (int o = 128; o > 0; o >>= 1) { if (t < o) red[t] = fmaxf(red[t], red[t + o]); __syncthreads(); }
  float mx = red[0]; __syncthreads();
  float e = expf(v - mx);
  red[t] = e; __syncthreads();
  for (int o = 128; o > 0; o >>= 1) { if (t < o) red[t] += red[t + o]; __syncthreads(); }
  row[t] = e / red[0];
}

__global__ __launch_bounds__(256)
void k_attn_renorm(const float* __restrict__ E, half_t* __restrict__ A16)
{
  __shared__ float red[256];
  const int col = blockIdx.x, t = threadIdx.x;   // col = b*256 + m ; t = n
  float v = E[(long long)t * 4096 + col];
  red[t] = v; __syncthreads();
  for (int o = 128; o > 0; o >>= 1) { if (t < o) red[t] += red[t + o]; __syncthreads(); }
  A16[(long long)t * 4096 + col] = (half_t)(v / (1e-9f + red[0]));
}

// per-batch 256x256 transpose: out[c, b*256+m] = in[m, b*256+c]
__global__ void k_transpose256(const half_t* __restrict__ in, half_t* __restrict__ out)
{
  long long i = (long long)blockIdx.x * 256 + threadIdx.x;
  if (i >= 256LL * 4096) return;
  int c = (int)(i / 4096);
  int col = (int)(i % 4096);
  int b = col >> 8, m = col & 255;
  out[i] = in[(long long)m * 4096 + b * 256 + c];
}

// g[b,c] = max_m X[c, b*256+m]
__global__ __launch_bounds__(256)
void k_colmax(const half_t* __restrict__ X, float* __restrict__ g)
{
  __shared__ float red[256];
  const int blk = blockIdx.x;          // b*1024 + c
  const int b = blk >> 10, c = blk & 1023, t = threadIdx.x;
  red[t] = (float)X[(long long)c * 4096 + b * 256 + t];
  __syncthreads();
  for (int o = 128; o > 0; o >>= 1) { if (t < o) red[t] = fmaxf(red[t], red[t + o]); __syncthreads(); }
  if (t == 0) g[blk] = red[0];
}

// ---------------------------------------------------------------------------
// Tiny FC head (B=16 rows — scalar is fine)
// ---------------------------------------------------------------------------
__global__ void k_fc(const float* __restrict__ in, const float* __restrict__ W,
                     float* __restrict__ out, int O, int I)
{
  int i = blockIdx.x * 256 + threadIdx.x;
  if (i >= BV * O) return;
  int b = i / O, o = i % O;
  float s = 0.f;
  for (int k = 0; k < I; ++k) s += in[b * I + k] * W[o * I + k];
  out[i] = s;
}

__global__ void k_bn_bc(float* __restrict__ X, const float* __restrict__ g,
                        const float* __restrict__ bb, int F)
{
  int f = blockIdx.x * 256 + threadIdx.x;
  if (f >= F) return;
  float m = 0.f;
  for (int b = 0; b < BV; ++b) m += X[b * F + f];
  m /= BV;
  float v = 0.f;
  for (int b = 0; b < BV; ++b) { float d = X[b * F + f] - m; v += d * d; }
  v /= BV;
  float sc = rsqrtf(v + 1e-5f) * g[f];
  for (int b = 0; b < BV; ++b) {
    float y = (X[b * F + f] - m) * sc + bb[f];
    X[b * F + f] = (y > 0.f) ? y : 0.2f * y;
  }
}

__global__ void k_final(const float* __restrict__ h, const float* __restrict__ w,
                        const float* __restrict__ b, float* __restrict__ out, int I)
{
  int bb_ = blockIdx.x * 256 + threadIdx.x;
  if (bb_ >= BV) return;
  float s = 0.f;
  for (int k = 0; k < I; ++k) s += h[bb_ * I + k] * w[k];
  out[bb_] = s + b[0];
}

// ===========================================================================
// Host orchestration
// ===========================================================================
extern "C" void kernel_launch(void* const* d_in, const int* in_sizes, int n_in,
                              void* d_out, int out_size, void* d_ws, size_t ws_size,
                              hipStream_t stream)
{
  (void)in_sizes; (void)n_in; (void)out_size; (void)ws_size;

  char* base = (char*)d_ws;
  size_t off = 0;
  auto alloc = [&](size_t bytes) -> void* {
    off = (off + 255) & ~size_t(255);
    void* p = base + off;
    off += bytes;
    return p;
  };
  auto allocH = [&](long long n) { return (half_t*)alloc((size_t)n * sizeof(half_t)); };
  auto allocF = [&](long long n) { return (float*)alloc((size_t)n * sizeof(float)); };
  auto allocI = [&](long long n) { return (int*)alloc((size_t)n * sizeof(int)); };

  auto P = [&](int i) { return (const float*)d_in[i]; };

  // ---- workspace layout ----
  half_t* w_conv1 = allocH(64 * 32);              // K padded 6->32
  half_t* w_conv2 = allocH(1024 * 64);
  half_t* w_l0w1 = allocH(128 * 128);
  half_t* w_l0w2 = allocH(128 * 128);
  half_t* w_l1w1 = allocH(256 * 256);
  half_t* w_l1w2 = allocH(256 * 256);
  half_t* w_fuse1 = allocH(256 * 1280);
  half_t* w_tr[6];  for (int i = 0; i < 6; ++i) w_tr[i] = allocH(256 * 256);
  half_t* w_qk[8];  for (int i = 0; i < 8; ++i) w_qk[i] = allocH(256 * 256);
  half_t* w_v[8];   for (int i = 0; i < 8; ++i) w_v[i]  = allocH(256 * 256);
  half_t* w_t[8];   for (int i = 0; i < 8; ++i) w_t[i]  = allocH(256 * 256);
  half_t* w_fuse2 = allocH(1024 * 1280);

  half_t* Ain   = allocH(32LL * BV * NV);         // (32, 131072), rows 6.. zero
  half_t* h16   = allocH(64LL * BV * NV);         // (64, 131072)
  half_t* hs16  = allocH(64LL * BV * 256);        // (64, 4096)
  half_t* xfuse = allocH(1280LL * 4096);          // rows 0-255 f1, 256-1279 x_str
  half_t* bufA  = allocH(128LL * 262144);         // group/GEMM ping
  half_t* bufB  = allocH(128LL * 262144);         // GEMM pong
  half_t* f0    = allocH(128LL * BV * 512);       // (128, 8192)
  half_t* xbuf  = allocH(256LL * 4096);
  half_t* xbuf2 = allocH(256LL * 4096);
  half_t* qb    = allocH(256LL * 4096);
  half_t* vb    = allocH(256LL * 4096);
  half_t* at16  = allocH(256LL * 4096);
  half_t* xr    = allocH(256LL * 4096);
  half_t* tm    = allocH(256LL * 4096);
  half_t* tb2   = allocH(256LL * 4096);
  half_t* so    = allocH(256LL * 4096);
  half_t* xp    = allocH(256LL * 4096);
  half_t* f2in  = allocH(1280LL * 4096);          // transformer outs + f1
  half_t* xg    = allocH(1024LL * 4096);

  float* xyz    = allocF(3LL * BV * NV);
  float* nxyz0  = allocF(3LL * BV * 512);
  float* nxyz1  = allocF(3LL * BV * 256);
  float* energy = allocF(256LL * 4096);
  float* d_mean = allocF(2048);
  float* d_var  = allocF(2048);
  float* gbuf   = allocF((long long)BV * 1024);
  float* h1buf  = allocF((long long)BV * 512);
  float* h2buf  = allocF((long long)BV * 256);

  int* idx0  = allocI(BV * 512);
  int* nidx0 = allocI((long long)BV * 512 * KNN);
  int* idx1  = allocI(BV * 256);
  int* nidx1 = allocI((long long)BV * 256 * KNN);

  // ---- helpers ----
  auto cvt = [&](int idx, half_t* dst, long long n) {
    k_f32_to_f16<<<dim3((unsigned)((n + 255) / 256)), dim3(256), 0, stream>>>(P(idx), dst, n);
  };
  auto gemm = [&](const half_t* A, const half_t* Bm, float* Cf, half_t* Ch,
                  int M, int N, int Kd, int lda, int ldb, int ldc,
                  int tA, int nb, long long sA, long long sB, long long sC) {
    dim3 g((N + TBN - 1) / TBN, (M + TBM - 1) / TBM, nb);
    k_gemm_wmma<<<g, dim3(256), 0, stream>>>(A, Bm, Cf, Ch, M, N, Kd, lda, ldb,
                                             ldc, tA, sA, sB, sC);
  };
  auto stats = [&](const half_t* X, int C, long long Pn) {
    k_rowstats<<<dim3(C), dim3(256), 0, stream>>>(X, d_mean, d_var, Pn);
  };
  auto bnact = [&](const half_t* X, const float* g, const float* b,
                   const half_t* res, half_t* dst, int C, long long Pn, int act) {
    long long tot = (long long)C * Pn;
    k_bn_act<<<dim3((unsigned)((tot + 255) / 256)), dim3(256), 0, stream>>>(
        X, d_mean, d_var, g, b, res, dst, Pn, C, act);
  };
  auto ew = [&](long long n) { return dim3((unsigned)((n + 255) / 256)); };

  // ---- convert weights ----
  k_pad_w<<<ew(64 * 32), dim3(256), 0, stream>>>(P(1), w_conv1, 64, 6, 32);
  cvt(4, w_conv2, 1024 * 64);
  cvt(7, w_l0w1, 128 * 128);  cvt(10, w_l0w2, 128 * 128);
  cvt(13, w_l1w1, 256 * 256); cvt(16, w_l1w2, 256 * 256);
  cvt(19, w_fuse1, 256 * 1280);
  for (int i = 0; i < 6; ++i) cvt(22 + i, w_tr[i], 256 * 256);
  for (int j = 0; j < 8; ++j) {
    cvt(40 + 7 * j + 0, w_qk[j], 256 * 256);
    cvt(40 + 7 * j + 1, w_v[j],  256 * 256);
    cvt(40 + 7 * j + 3, w_t[j],  256 * 256);
  }
  cvt(96, w_fuse2, 1024 * 1280);

  // ---- input pack + conv1/bn1/relu -> h (64, B*N) ----
  const long long P0 = (long long)BV * NV;   // 131072
  k_pack_input<<<ew(P0), dim3(256), 0, stream>>>((const float*)d_in[0], xyz, Ain);
  gemm(w_conv1, Ain, nullptr, h16, 64, (int)P0, 32, 32, (int)P0, (int)P0, 0, 1, 0, 0, 0);
  stats(h16, 64, P0);
  bnact(h16, P(2), P(3), nullptr, h16, 64, P0, 1);

  // ---- x_str: strided sample + conv2/bn2/relu directly into xfuse rows 256.. ----
  k_stride_cols<<<ew(64LL * BV * 256), dim3(256), 0, stream>>>(h16, hs16);
  half_t* xs = xfuse + 256LL * 4096;
  gemm(w_conv2, hs16, nullptr, xs, 1024, 4096, 64, 64, 4096, 4096, 0, 1, 0, 0, 0);
  stats(xs, 1024, 4096);
  bnact(xs, P(5), P(6), nullptr, xs, 1024, 4096, 1);

  // ---- stage 0: FPS(512) + kNN + group + local_op -> f0 (128, 8192) ----
  k_fps<<<dim3(BV), dim3(256), 0, stream>>>(xyz, idx0, NV, 512);
  k_gather_xyz<<<ew(BV * 512), dim3(256), 0, stream>>>(xyz, idx0, nxyz0, NV, 512);
  k_knn<<<dim3(BV * 512), dim3(256), 0, stream>>>(xyz, nxyz0, nidx0, NV, 512, KNN);
  const long long P1 = (long long)BV * 512 * KNN;  // 262144
  k_group<<<ew(P1), dim3(256), 0, stream>>>(h16, idx0, nidx0, bufA, 64, NV, 512, KNN);
  gemm(w_l0w1, bufA, nullptr, bufB, 128, (int)P1, 128, 128, (int)P1, (int)P1, 0, 1, 0, 0, 0);
  stats(bufB, 128, P1); bnact(bufB, P(8), P(9), nullptr, bufB, 128, P1, 1);
  gemm(w_l0w2, bufB, nullptr, bufA, 128, (int)P1, 128, 128, (int)P1, (int)P1, 0, 1, 0, 0, 0);
  stats(bufA, 128, P1); bnact(bufA, P(11), P(12), nullptr, bufA, 128, P1, 1);
  k_maxpool<<<ew(128LL * BV * 512), dim3(256), 0, stream>>>(bufA, f0, (long long)BV * 512, KNN, 128);

  // ---- stage 1: FPS(256) on nxyz0 + group over f0 + local_op -> f1 (xfuse rows 0-255) ----
  k_fps<<<dim3(BV), dim3(256), 0, stream>>>(nxyz0, idx1, 512, 256);
  k_gather_xyz<<<ew(BV * 256), dim3(256), 0, stream>>>(nxyz0, idx1, nxyz1, 512, 256);
  k_knn<<<dim3(BV * 256), dim3(256), 0, stream>>>(nxyz0, nxyz1, nidx1, 512, 256, KNN);
  const long long P2 = (long long)BV * 256 * KNN;  // 131072
  k_group<<<ew(P2), dim3(256), 0, stream>>>(f0, idx1, nidx1, bufA, 128, 512, 256, KNN);
  gemm(w_l1w1, bufA, nullptr, bufB, 256, (int)P2, 256, 256, (int)P2, (int)P2, 0, 1, 0, 0, 0);
  stats(bufB, 256, P2); bnact(bufB, P(14), P(15), nullptr, bufB, 256, P2, 1);
  gemm(w_l1w2, bufB, nullptr, bufA, 256, (int)P2, 256, 256, (int)P2, (int)P2, 0, 1, 0, 0, 0);
  stats(bufA, 256, P2); bnact(bufA, P(17), P(18), nullptr, bufA, 256, P2, 1);
  k_maxpool<<<ew(256LL * 4096), dim3(256), 0, stream>>>(bufA, xfuse, 4096, KNN, 256);

  // ---- fuse1: (256,1280) x (1280,4096), BN + leaky -> x (xbuf) ----
  gemm(w_fuse1, xfuse, nullptr, xbuf, 256, 4096, 1280, 1280, 4096, 4096, 0, 1, 0, 0, 0);
  stats(xbuf, 256, 4096); bnact(xbuf, P(20), P(21), nullptr, xbuf, 256, 4096, 2);

  // ---- transformer head convs w[0], w[1] ----
  gemm(w_tr[0], xbuf, nullptr, xbuf2, 256, 4096, 256, 256, 4096, 4096, 0, 1, 0, 0, 0);
  stats(xbuf2, 256, 4096); bnact(xbuf2, P(28), P(34), nullptr, xbuf2, 256, 4096, 1);
  gemm(w_tr[1], xbuf2, nullptr, xbuf, 256, 4096, 256, 256, 4096, 4096, 0, 1, 0, 0, 0);
  stats(xbuf, 256, 4096); bnact(xbuf, P(29), P(35), nullptr, xbuf, 256, 4096, 1);

  const long long E16 = 256LL * 4096;
  auto run_sa = [&](const half_t* xin, int j, half_t* out) {
    const int pb = 40 + 7 * j;
    gemm(w_qk[j], xin, nullptr, qb, 256, 4096, 256, 256, 4096, 4096, 0, 1, 0, 0, 0);
    gemm(w_v[j],  xin, nullptr, vb, 256, 4096, 256, 256, 4096, 4096, 0, 1, 0, 0, 0);
    k_bias<<<ew(E16), dim3(256), 0, stream>>>(vb, P(pb + 2), 4096, 256);
    // energy[b](n,m) = sum_c q[b,c,n] q[b,c,m]   (per-batch column blocks)
    gemm(qb, qb, energy, nullptr, 256, 256, 256, 4096, 4096, 4096, 1, BV, 256, 256, 256);
    k_softmax_rows<<<dim3(BV * 256), dim3(256), 0, stream>>>(energy);
    k_attn_renorm<<<dim3(4096), dim3(256), 0, stream>>>(energy, at16);
    // x_r[b](c,m) = sum_n v[b,c,n] attn[b,n,m]
    gemm(vb, at16, nullptr, xr, 256, 256, 256, 4096, 4096, 4096, 0, BV, 256, 256, 256);
    k_sub<<<ew(E16), dim3(256), 0, stream>>>(tm, xin, xr, E16);
    gemm(w_t[j], tm, nullptr, tb2, 256, 4096, 256, 256, 4096, 4096, 0, 1, 0, 0, 0);
    // t_b cancels inside training-mode BN (per-channel constant); skip adding it.
    stats(tb2, 256, 4096);
    bnact(tb2, P(pb + 5), P(pb + 6), xin, out, 256, 4096, 1);  // out = x + relu(bn(t))
  };

  half_t* x = xbuf;
  for (int i = 0; i < 4; ++i) {
    run_sa(x, 2 * i, so);
    k_add<<<ew(E16), dim3(256), 0, stream>>>(qb, so, x, E16);     // xa = sa(x)+x
    k_transpose256<<<ew(E16), dim3(256), 0, stream>>>(qb, xp);    // xp = xa^T (per batch)
    run_sa(xp, 2 * i + 1, so);
    k_add<<<ew(E16), dim3(256), 0, stream>>>(tm, so, xp, E16);    // xb = sa(xp)+xp
    half_t* dst = f2in + (long long)(256 * i) * 4096;
    gemm(w_tr[2 + i], tm, nullptr, dst, 256, 4096, 256, 256, 4096, 4096, 0, 1, 0, 0, 0);
    stats(dst, 256, 4096);
    bnact(dst, P(30 + i), P(36 + i), nullptr, dst, 256, 4096, 1);
    x = dst;
  }

  // ---- fuse2: concat(transformer outs, f1) -> (1024, 4096), BN + leaky ----
  k_copy_h<<<ew(256LL * 4096), dim3(256), 0, stream>>>(f2in + 1024LL * 4096, xfuse, 256LL * 4096);
  gemm(w_fuse2, f2in, nullptr, xg, 1024, 4096, 1280, 1280, 4096, 4096, 0, 1, 0, 0, 0);
  stats(xg, 1024, 4096); bnact(xg, P(97), P(98), nullptr, xg, 1024, 4096, 2);

  // ---- global max pool + FC head ----
  k_colmax<<<dim3(BV * 1024), dim3(256), 0, stream>>>(xg, gbuf);
  k_fc<<<ew(BV * 512), dim3(256), 0, stream>>>(gbuf, P(99), h1buf, 512, 1024);
  k_bn_bc<<<ew(512), dim3(256), 0, stream>>>(h1buf, P(100), P(101), 512);
  k_fc<<<ew(BV * 256), dim3(256), 0, stream>>>(h1buf, P(102), h2buf, 256, 512);
  k_bn_bc<<<ew(256), dim3(256), 0, stream>>>(h2buf, P(103), P(104), 256);
  k_final<<<dim3(1), dim3(256), 0, stream>>>(h2buf, P(105), P(106), (float*)d_out, 256);
}
